// TransformerLanguageModel_16767552324252
// MI455X (gfx1250) — matrix operand order repaired and checked by reading, not compiled, not run on hardware
//
#include <hip/hip_runtime.h>
#include <hip/hip_bf16.h>

// ---- problem dims ----
#define NB   2
#define NT   2048
#define ND   512
#define NH   8
#define NL   4
#define NVOC 50256
#define NHS  64
#define NROWS (NB*NT)          // 4096
#define NEPS 1e-5f

typedef __attribute__((ext_vector_type(16))) __bf16 v16bf;
typedef __attribute__((ext_vector_type(8)))  float  v8f;

union FragB { v16bf v; unsigned int u[8]; unsigned short s[16]; };
union U4    { uint4 v; unsigned int u[4]; unsigned short s[8]; };

__device__ __forceinline__ unsigned short f2bf(float f) {
  union { float f; unsigned int u; } c; c.f = f;
  unsigned int r = c.u + 0x7FFFu + ((c.u >> 16) & 1u);
  return (unsigned short)(r >> 16);
}

// A-matrix (16x32 bf16) uint-pair index within a 16-uint k-chunk, per ISA 7.12.2
__device__ __forceinline__ int amapu(int p, int half) {
  return (p < 4) ? (p + 4*half) : (4 + p + 4*half);  // = 8+(p-4)+4*half for p>=4
}

__device__ __forceinline__ v8f wmma_bf16(v16bf a, v16bf b, v8f c) {
  return __builtin_amdgcn_wmma_f32_16x16x32_bf16(false, a, false, b, (short)0, c,
                                                 false, false);
}

// per-lane 16B global->LDS async copy (CDNA5 TDM-adjacent async path, ASYNCcnt)
__device__ __forceinline__ void async_g2l_b128(unsigned lds_addr, const void* gaddr) {
  asm volatile("global_load_async_to_lds_b128 %0, %1, off"
               :: "v"(lds_addr), "v"((unsigned long long)(size_t)gaddr) : "memory");
}
__device__ __forceinline__ void wait_async0() {
  asm volatile("s_wait_asynccnt 0x0" ::: "memory");
}

// ---------------- fp32 -> bf16 weight conversion ----------------
__global__ void cvt_bf16_kernel(const float* __restrict__ src,
                                unsigned short* __restrict__ dst, long n) {
  long i = (long)blockIdx.x * blockDim.x + threadIdx.x;
  if (i < n) dst[i] = f2bf(src[i]);
}

// ---------------- embedding + sinusoidal positional encoding ----------------
__global__ void embed_kernel(const int* __restrict__ idx,
                             const float* __restrict__ emb,
                             float* __restrict__ x) {
  int row = blockIdx.x;               // 0..NROWS-1
  int t   = row % NT;
  int tok = idx[row];
  const float* e = emb + (long)tok * ND;
  const float LOG1E4 = 9.210340371976184f;
  for (int d = threadIdx.x; d < ND; d += 256) {
    int d2 = d & ~1;
    float freq = __expf(-LOG1E4 * (float)d2 / (float)ND);
    float ang  = (float)t * freq;
    float pe   = (d & 1) ? __cosf(ang) : __sinf(ang);
    x[(long)row * ND + d] = e[d] + pe;
  }
}

// ---------------- LayerNorm (f32 in -> bf16 out) ----------------
__global__ __launch_bounds__(256)
void ln_kernel(const float* __restrict__ x, const float* __restrict__ g,
               const float* __restrict__ b, unsigned short* __restrict__ h) {
  __shared__ float red[8];
  int row = blockIdx.x;
  const float* xr = x + (long)row * ND;
  int wave = threadIdx.x >> 5, lane = threadIdx.x & 31;

  float s = 0.f;
  for (int d = threadIdx.x; d < ND; d += 256) s += xr[d];
  for (int o = 16; o >= 1; o >>= 1) s += __shfl_xor(s, o, 32);
  if (lane == 0) red[wave] = s;
  __syncthreads();
  float tot = 0.f;
  for (int w = 0; w < 8; w++) tot += red[w];
  float mu = tot / (float)ND;
  __syncthreads();

  float v = 0.f;
  for (int d = threadIdx.x; d < ND; d += 256) { float df = xr[d] - mu; v += df * df; }
  for (int o = 16; o >= 1; o >>= 1) v += __shfl_xor(v, o, 32);
  if (lane == 0) red[wave] = v;
  __syncthreads();
  float vt = 0.f;
  for (int w = 0; w < 8; w++) vt += red[w];
  float inv = rsqrtf(vt / (float)ND + NEPS);

  for (int d = threadIdx.x; d < ND; d += 256)
    h[(long)row * ND + d] = f2bf((xr[d] - mu) * inv * g[d] + b[d]);
}

// ---------------- bf16 WMMA GEMM: C = A[M,K] * B[K,N] (+bias)(relu)(+resid) ---
// block = 256 thr (8 waves, 4x2), tile 128x64, K-tile 64, wave = 32x32 output.
// A tile arrives via global_load_async_to_lds_b128 (ASYNCcnt), B tile is
// transposed into LDS so B fragments are contiguous ds_load_b128 pairs.
__global__ __launch_bounds__(256)
void gemm_kernel(const unsigned short* __restrict__ A,
                 const unsigned short* __restrict__ Bm,
                 const float* __restrict__ bias,
                 const float* __restrict__ resid,
                 float* __restrict__ outF,
                 unsigned short* __restrict__ outB,
                 int N, int K, int relu) {
  __shared__ __align__(16) unsigned int ldsA[128 * 32];  // 128 rows x 64 bf16 (16KB)
  __shared__ __align__(16) unsigned int ldsB[64 * 32];   // 64 cols x 64 k, transposed (8KB)

  int tid = threadIdx.x;
  int wave = tid >> 5, lane = tid & 31;
  int wm = wave >> 1, wn = wave & 1;
  int m0 = blockIdx.y * 128;
  int n0 = blockIdx.x * 64;
  int half = lane >> 4, l15 = lane & 15;
  unsigned ldsA_base = (unsigned)(size_t)ldsA;  // low 32 bits of flat = LDS offset

  v8f acc[2][2] = {};

  for (int k0 = 0; k0 < K; k0 += 64) {
    __syncthreads();
    // ---- A tile: 128x64 bf16 = 1024 x 16B async copies, 4 per thread ----
    for (int i = 0; i < 4; i++) {
      int u = tid + i * 256;              // 0..1023
      int row = u >> 3, cu = u & 7;
      const unsigned short* g = A + (size_t)(m0 + row) * K + k0 + cu * 8;
      async_g2l_b128(ldsA_base + (unsigned)u * 16u, g);
    }
    // ---- B tile: 64x64 bf16 loaded row-major, stored transposed ----
    for (int i = 0; i < 2; i++) {
      int slot = tid + i * 256;           // 0..511
      int kk = slot >> 3;                 // 0..63
      int c8 = (slot & 7) << 3;           // 0..56
      U4 u4;
      if (n0 + c8 + 8 <= N) {
        u4.v = *(const uint4*)(Bm + (size_t)(k0 + kk) * N + n0 + c8);
      } else {
        for (int j2 = 0; j2 < 4; j2++) u4.u[j2] = 0u;
      }
      unsigned short* lb = (unsigned short*)ldsB;
      for (int j2 = 0; j2 < 8; j2++) lb[(c8 + j2) * 64 + kk] = u4.s[j2];
    }
    if (k0 + 64 < K) {
      __builtin_prefetch(Bm + (size_t)(k0 + 64 + (tid >> 3)) * N + n0 + ((tid & 7) << 3), 0, 3);
    }
    wait_async0();          // our A copies are in LDS
    __syncthreads();        // (compiler drains DScnt for B stores at the barrier)

    // ---- 2 k-chunks x 4 WMMA ----
    for (int c = 0; c < 2; c++) {
      FragB a0, a1, b0, b1;
      for (int p = 0; p < 8; p++) {
        int au = c * 16 + amapu(p, half);
        a0.u[p] = ldsA[(wm * 32 + l15) * 32 + au];
        a1.u[p] = ldsA[(wm * 32 + 16 + l15) * 32 + au];
        int bu = c * 16 + amapu(p, half);
        b0.u[p] = ldsB[(wn * 32 + l15) * 32 + bu];
        b1.u[p] = ldsB[(wn * 32 + 16 + l15) * 32 + bu];
      }
      acc[0][0] = wmma_bf16(a0.v, b0.v, acc[0][0]);
      acc[0][1] = wmma_bf16(a0.v, b1.v, acc[0][1]);
      acc[1][0] = wmma_bf16(a1.v, b0.v, acc[1][0]);
      acc[1][1] = wmma_bf16(a1.v, b1.v, acc[1][1]);
    }
  }

  // ---- epilogue ----
  for (int ti = 0; ti < 2; ti++)
    for (int tj = 0; tj < 2; tj++)
      for (int r = 0; r < 8; r++) {
        int row = m0 + wm * 32 + ti * 16 + r + 8 * half;
        int col = n0 + wn * 32 + tj * 16 + l15;
        if (col < N) {
          float v = acc[ti][tj][r];
          if (bias)  v += bias[col];
          if (relu)  v = fmaxf(v, 0.f);
          if (resid) v += resid[(size_t)row * N + col];
          if (outF)  outF[(size_t)row * N + col] = v;
          if (outB)  outB[(size_t)row * N + col] = f2bf(v);
        }
      }
}

// ---------------- flash attention, 1 wave per (b,h,16-query tile) -----------
// V tiles stream into wave-private LDS with async copies, issued before the
// QK^T WMMAs so the copy overlaps score computation and softmax.
__global__ __launch_bounds__(128)
void attn_kernel(const unsigned short* __restrict__ qb,
                 const unsigned short* __restrict__ kb,
                 const unsigned short* __restrict__ vb,
                 unsigned short* __restrict__ ob) {
  __shared__ unsigned int ldsP[4][256];        // per wave: 16 rows x 32 bf16 (1KB)
  __shared__ __align__(16) unsigned int ldsV[4][1024];  // per wave: 32 x 64 bf16 (4KB)
  int wave = threadIdx.x >> 5, lane = threadIdx.x & 31;
  int tile = blockIdx.x * 4 + wave;            // 0..2047
  const int T16 = NT / 16;
  int b  = tile / (NH * T16);
  int rm = tile % (NH * T16);
  int h  = rm / T16;
  int qt = rm % T16;
  int qbase = qt * 16;
  int half = lane >> 4, l15 = lane & 15;
  unsigned vb_lds = (unsigned)(size_t)(&ldsV[wave][0]);
  const unsigned short* lv = (const unsigned short*)(&ldsV[wave][0]);

  // Q fragments (16x64 -> two 16x32 bf16 A-frags)
  FragB aq[2];
  {
    const unsigned int* qrow =
        (const unsigned int*)(qb + (size_t)(b * NT + qbase + l15) * ND + h * NHS);
    for (int c = 0; c < 2; c++)
      for (int p = 0; p < 8; p++)
        aq[c].u[p] = qrow[c * 16 + amapu(p, half)];
  }

  v8f o0 = {}, o1 = {}, o2 = {}, o3 = {};
  float m8[8], l8[8];
  for (int r = 0; r < 8; r++) { m8[r] = -1e30f; l8[r] = 0.f; }

  const float sc = 0.125f;                // 1/sqrt(64)
  int kend = qbase + 16;
  for (int kb0 = 0; kb0 < kend; kb0 += 32) {
    // ---- kick off async V tile (32 keys x 64 dims) into wave-private LDS ----
    for (int q = 0; q < 8; q++) {
      int slot = q * 32 + lane;           // 0..255 (16B granules)
      int row = slot >> 3, c4 = slot & 7;
      const unsigned short* g =
          vb + (size_t)(b * NT + kb0 + row) * ND + h * NHS + c4 * 8;
      async_g2l_b128(vb_lds + (unsigned)slot * 16u, g);
    }
    // ---- S = Q K^T for 32 keys (two 16x16 tiles) ----
    v8f s0 = {}, s1 = {};
    for (int ct = 0; ct < 2; ct++) {
      const unsigned int* krow =
          (const unsigned int*)(kb + (size_t)(b * NT + kb0 + ct * 16 + l15) * ND + h * NHS);
      v8f s = {};
      for (int c = 0; c < 2; c++) {
        FragB bf;
        for (int p = 0; p < 8; p++) bf.u[p] = krow[c * 16 + amapu(p, half)];
        s = wmma_bf16(aq[c].v, bf.v, s);
      }
      if (ct == 0) s0 = s; else s1 = s;
    }
    // ---- scale + causal mask ----
    for (int r = 0; r < 8; r++) {
      int q = qbase + r + 8 * half;
      float v0 = s0[r] * sc; if (kb0 + l15 > q)      v0 = -1e30f;
      float v1 = s1[r] * sc; if (kb0 + 16 + l15 > q) v1 = -1e30f;
      s0[r] = v0; s1[r] = v1;
    }
    // ---- online softmax (row reductions across 16-lane half groups) ----
    float p0[8], p1[8];
    for (int r = 0; r < 8; r++) {
      float mx = fmaxf(s0[r], s1[r]);
      for (int o = 8; o >= 1; o >>= 1) mx = fmaxf(mx, __shfl_xor(mx, o, 32));
      float nm = fmaxf(m8[r], mx);
      float e0 = __expf(s0[r] - nm);
      float e1 = __expf(s1[r] - nm);
      float rs = e0 + e1;
      for (int o = 8; o >= 1; o >>= 1) rs += __shfl_xor(rs, o, 32);
      float alpha = __expf(m8[r] - nm);
      l8[r] = l8[r] * alpha + rs;
      m8[r] = nm;
      p0[r] = e0; p1[r] = e1;
      o0[r] *= alpha; o1[r] *= alpha; o2[r] *= alpha; o3[r] *= alpha;
    }
    // ---- stage P (16x32 bf16) through wave-private LDS -> A fragment ----
    {
      unsigned short* Pl = (unsigned short*)ldsP[wave];
      for (int r = 0; r < 8; r++) {
        Pl[(r + 8 * half) * 32 + l15]      = f2bf(p0[r]);
        Pl[(r + 8 * half) * 32 + 16 + l15] = f2bf(p1[r]);
      }
    }
    asm volatile("s_wait_dscnt 0" ::: "memory");
    FragB aP;
    for (int p = 0; p < 8; p++) aP.u[p] = ldsP[wave][l15 * 16 + p + 8 * half];
    // ---- O += P * V (32 keys x 64 dims -> 4 B-frags from LDS V tile) ----
    wait_async0();   // V tile resident
    for (int nt = 0; nt < 4; nt++) {
      FragB bv;
      for (int j = 0; j < 16; j++) {
        int kk = j + 16 * half;
        bv.s[j] = lv[kk * 64 + nt * 16 + l15];
      }
      v8f* op = (nt == 0) ? &o0 : (nt == 1) ? &o1 : (nt == 2) ? &o2 : &o3;
      *op = wmma_bf16(aP.v, bv.v, *op);
    }
  }
  // ---- finalize ----
  for (int r = 0; r < 8; r++) {
    float inv = 1.f / l8[r];
    size_t row = (size_t)(b * NT + qbase + r + 8 * half);
    ob[row * ND + h * NHS +  0 + l15] = f2bf(o0[r] * inv);
    ob[row * ND + h * NHS + 16 + l15] = f2bf(o1[r] * inv);
    ob[row * ND + h * NHS + 32 + l15] = f2bf(o2[r] * inv);
    ob[row * ND + h * NHS + 48 + l15] = f2bf(o3[r] * inv);
  }
}

// ---------------- loss: per-row logsumexp + nll, mean via atomics ------------
__global__ void zero_loss_kernel(float* loss) { *loss = 0.f; }

__global__ __launch_bounds__(256)
void loss_kernel(const float* __restrict__ logits, const int* __restrict__ targets,
                 float* __restrict__ loss) {
  __shared__ float red[8];
  int row = blockIdx.x;
  const float* lr = logits + (size_t)row * NVOC;
  int wave = threadIdx.x >> 5, lane = threadIdx.x & 31;

  float mx = -1e30f;
  for (int i = threadIdx.x; i < NVOC; i += 256) mx = fmaxf(mx, lr[i]);
  for (int o = 16; o >= 1; o >>= 1) mx = fmaxf(mx, __shfl_xor(mx, o, 32));
  if (lane == 0) red[wave] = mx;
  __syncthreads();
  float bm = red[0];
  for (int w = 1; w < 8; w++) bm = fmaxf(bm, red[w]);
  __syncthreads();

  float s = 0.f;
  for (int i = threadIdx.x; i < NVOC; i += 256) s += __expf(lr[i] - bm);
  for (int o = 16; o >= 1; o >>= 1) s += __shfl_xor(s, o, 32);
  if (lane == 0) red[wave] = s;
  __syncthreads();
  if (threadIdx.x == 0) {
    float tot = 0.f;
    for (int w = 0; w < 8; w++) tot += red[w];
    float nll = logf(tot) + bm - lr[targets[row]];
    atomicAdd(loss, nll / (float)NROWS);
  }
}

// ---------------- host orchestration ----------------
extern "C" void kernel_launch(void* const* d_in, const int* in_sizes, int n_in,
                              void* d_out, int out_size, void* d_ws, size_t ws_size,
                              hipStream_t stream) {
  (void)in_sizes; (void)n_in; (void)out_size; (void)ws_size;
  const int*   idx     = (const int*)d_in[0];
  const int*   targets = (const int*)d_in[1];
  const float* emb     = (const float*)d_in[2];
  const float* Wq      = (const float*)d_in[3];
  const float* Wk      = (const float*)d_in[4];
  const float* Wv      = (const float*)d_in[5];
  const float* Wo      = (const float*)d_in[6];
  const float* bo      = (const float*)d_in[7];
  const float* ln1_g   = (const float*)d_in[8];
  const float* ln1_b   = (const float*)d_in[9];
  const float* ln2_g   = (const float*)d_in[10];
  const float* ln2_b   = (const float*)d_in[11];
  const float* W1      = (const float*)d_in[12];
  const float* b1      = (const float*)d_in[13];
  const float* W2      = (const float*)d_in[14];
  const float* b2      = (const float*)d_in[15];
  const float* lnf_g   = (const float*)d_in[16];
  const float* lnf_b   = (const float*)d_in[17];
  const float* Wout    = (const float*)d_in[18];
  const float* bout    = (const float*)d_in[19];

  // workspace carve-up (256B aligned chunks)
  char* wp = (char*)d_ws;
  auto alloc = [&](size_t bytes) -> void* {
    void* p = (void*)wp;
    wp += (bytes + 255) & ~(size_t)255;
    return p;
  };
  const size_t szW  = (size_t)NL * ND * ND;       // per Wq/Wk/Wv/Wo
  const size_t szW1 = (size_t)NL * ND * 4 * ND;
  const size_t szWo = (size_t)ND * NVOC;

  unsigned short* wq_b = (unsigned short*)alloc(szW  * 2);
  unsigned short* wk_b = (unsigned short*)alloc(szW  * 2);
  unsigned short* wv_b = (unsigned short*)alloc(szW  * 2);
  unsigned short* wo_b = (unsigned short*)alloc(szW  * 2);
  unsigned short* w1_b = (unsigned short*)alloc(szW1 * 2);
  unsigned short* w2_b = (unsigned short*)alloc(szW1 * 2);
  unsigned short* wt_b = (unsigned short*)alloc(szWo * 2);
  float*          x    = (float*)         alloc((size_t)NROWS * ND * 4);
  unsigned short* hb   = (unsigned short*)alloc((size_t)NROWS * ND * 2);
  unsigned short* qbuf = (unsigned short*)alloc((size_t)NROWS * ND * 2);
  unsigned short* kbuf = (unsigned short*)alloc((size_t)NROWS * ND * 2);
  unsigned short* vbuf = (unsigned short*)alloc((size_t)NROWS * ND * 2);
  unsigned short* aout = (unsigned short*)alloc((size_t)NROWS * ND * 2);
  unsigned short* mlp  = (unsigned short*)alloc((size_t)NROWS * 4 * ND * 2);

  auto cvt = [&](const float* s, unsigned short* d, size_t n) {
    cvt_bf16_kernel<<<(unsigned)((n + 255) / 256), 256, 0, stream>>>(s, d, (long)n);
  };
  cvt(Wq, wq_b, szW);  cvt(Wk, wk_b, szW);  cvt(Wv, wv_b, szW);  cvt(Wo, wo_b, szW);
  cvt(W1, w1_b, szW1); cvt(W2, w2_b, szW1); cvt(Wout, wt_b, szWo);

  embed_kernel<<<NROWS, 256, 0, stream>>>(idx, emb, x);

  dim3 gD(ND / 64, NROWS / 128);        // (8, 32)
  dim3 g4D(4 * ND / 64, NROWS / 128);   // (32, 32)
  dim3 gV((NVOC + 63) / 64, NROWS / 128);

  for (int l = 0; l < NL; l++) {
    ln_kernel<<<NROWS, 256, 0, stream>>>(x, ln1_g + l * ND, ln1_b + l * ND, hb);
    gemm_kernel<<<gD, 256, 0, stream>>>(hb, wq_b + (size_t)l * ND * ND,
                                        nullptr, nullptr, nullptr, qbuf, ND, ND, 0);
    gemm_kernel<<<gD, 256, 0, stream>>>(hb, wk_b + (size_t)l * ND * ND,
                                        nullptr, nullptr, nullptr, kbuf, ND, ND, 0);
    gemm_kernel<<<gD, 256, 0, stream>>>(hb, wv_b + (size_t)l * ND * ND,
                                        nullptr, nullptr, nullptr, vbuf, ND, ND, 0);
    attn_kernel<<<(NB * NH * (NT / 16)) / 4, 128, 0, stream>>>(qbuf, kbuf, vbuf, aout);
    gemm_kernel<<<gD, 256, 0, stream>>>(aout, wo_b + (size_t)l * ND * ND,
                                        bo + l * ND, x, x, nullptr, ND, ND, 0);
    ln_kernel<<<NROWS, 256, 0, stream>>>(x, ln2_g + l * ND, ln2_b + l * ND, hb);
    gemm_kernel<<<g4D, 256, 0, stream>>>(hb, w1_b + (size_t)l * ND * 4 * ND,
                                         b1 + l * 4 * ND, nullptr, nullptr, mlp,
                                         4 * ND, ND, 1);
    gemm_kernel<<<gD, 256, 0, stream>>>(mlp, w2_b + (size_t)l * 4 * ND * ND,
                                        b2 + l * ND, x, x, nullptr, ND, 4 * ND, 0);
  }

  ln_kernel<<<NROWS, 256, 0, stream>>>(x, lnf_g, lnf_b, hb);
  float* logits = (float*)d_out;
  gemm_kernel<<<gV, 256, 0, stream>>>(hb, wt_b, bout, nullptr, logits, nullptr,
                                      NVOC, ND, 0);
  float* loss = logits + (size_t)NROWS * NVOC;
  zero_loss_kernel<<<1, 1, 0, stream>>>(loss);
  loss_kernel<<<NROWS, 256, 0, stream>>>(logits, targets, loss);
}
